// DoubleStreamBlockWrapper_38474317038520
// MI455X (gfx1250) — matrix operand order repaired and checked by reading, not compiled, not run on hardware
//
#include <hip/hip_runtime.h>
#include <cstddef>

// ---------------- problem constants ----------------
#define BATCH   1
#define L_TXT   256
#define L_IMG   2048
#define L_LIVE  2304
#define L_FULL  3072
#define HID     2048
#define NH      16
#define HD      128
#define MLPD    8192

typedef __attribute__((ext_vector_type(16))) __bf16 v16b;
typedef __attribute__((ext_vector_type(8)))  float  v8f;

#define WMMA_BF16(a, b, c) \
  __builtin_amdgcn_wmma_f32_16x16x32_bf16(false, (a), false, (b), (short)0, (c), false, false)

// ---------------- bf16 helpers ----------------
__device__ __forceinline__ __bf16 f2b(float f) {
  unsigned u = __builtin_bit_cast(unsigned, f);
  unsigned r = (u + 0x7FFFu + ((u >> 16) & 1u)) >> 16;
  return __builtin_bit_cast(__bf16, (unsigned short)r);
}
__device__ __forceinline__ float b2f(__bf16 h) {
  unsigned u = ((unsigned)__builtin_bit_cast(unsigned short, h)) << 16;
  return __builtin_bit_cast(float, u);
}

// ---------------- DPP16 cross-lane reductions (pure VALU, no DS traffic) ----
// Butterfly masks {1,2,7,15} span {1,2,4,8}: full reduction within each
// 16-lane row.  quad_perm xor1 = 0xB1, quad_perm xor2 = 0x4E,
// row_half_mirror = 0x141 (xor7), row_mirror = 0x140 (xor15).
template <int CTRL>
__device__ __forceinline__ float dpp_mov(float x) {
  return __builtin_bit_cast(
      float, __builtin_amdgcn_update_dpp(0, __builtin_bit_cast(int, x), CTRL, 0xF, 0xF, true));
}
__device__ __forceinline__ float row_max16(float x) {
  x = fmaxf(x, dpp_mov<0xB1>(x));
  x = fmaxf(x, dpp_mov<0x4E>(x));
  x = fmaxf(x, dpp_mov<0x141>(x));
  x = fmaxf(x, dpp_mov<0x140>(x));
  return x;
}
__device__ __forceinline__ float row_sum16(float x) {
  x += dpp_mov<0xB1>(x);
  x += dpp_mov<0x4E>(x);
  x += dpp_mov<0x141>(x);
  x += dpp_mov<0x140>(x);
  return x;
}
// Full-wave (32-lane) sum: DPP within rows + one cross-row shuffle.
__device__ __forceinline__ float wave_sum32(float x) {
  x = row_sum16(x);
  x += __shfl_xor(x, 16, 32);
  return x;
}

// ---------------- WMMA fragment loaders (contiguous k = 16*half + s, same order for A and B) ----------------
// A operand, 16x32 bf16, source row-major rows of length ld.
// lane m = lane&15 is the row; within a 32-wide k chunk:
//   k(v) = 16*half + 2v, pairs contiguous -> 32-bit loads.
__device__ __forceinline__ v16b frag_a_bf16(const __bf16* A, int ld, int m, int hf, int k0) {
  v16b f;
  const __bf16* row = A + (size_t)m * ld + k0 + 16 * hf;
#pragma unroll
  for (int v = 0; v < 8; ++v) {
    int k = 2 * v;
    unsigned u = *reinterpret_cast<const unsigned*>(row + k);
    f[2 * v]     = __builtin_bit_cast(__bf16, (unsigned short)(u & 0xFFFFu));
    f[2 * v + 1] = __builtin_bit_cast(__bf16, (unsigned short)(u >> 16));
  }
  return f;
}

// B operand, 32x16.  B[k][n] = Src[n][k]  (source rows over n, k contiguous):
// lane n = lane&15; k(v) = 16*half + 2v  -> 32-bit loads.  bf16 source.
__device__ __forceinline__ v16b frag_b_nk_bf16(const __bf16* Src, int ld, int n, int hf, int k0) {
  v16b f;
  const __bf16* row = Src + (size_t)n * ld + k0 + 16 * hf;
#pragma unroll
  for (int v = 0; v < 8; ++v) {
    unsigned u = *reinterpret_cast<const unsigned*>(row + 2 * v);
    f[2 * v]     = __builtin_bit_cast(__bf16, (unsigned short)(u & 0xFFFFu));
    f[2 * v + 1] = __builtin_bit_cast(__bf16, (unsigned short)(u >> 16));
  }
  return f;
}

// Same, but source is fp32 weights (N x K row-major); convert on load.
__device__ __forceinline__ v16b frag_b_nk_f32(const float* W, int ldk, int n, int hf, int k0) {
  v16b f;
  const float* row = W + (size_t)n * ldk + k0 + 16 * hf;
#pragma unroll
  for (int v = 0; v < 8; ++v) {
    float2 p = *reinterpret_cast<const float2*>(row + 2 * v);
    f[2 * v]     = f2b(p.x);
    f[2 * v + 1] = f2b(p.y);
  }
  return f;
}

// B operand where source is stored [k][n] (the V matrix): strided 16-bit loads.
__device__ __forceinline__ v16b frag_b_kn_bf16(const __bf16* Src, int ld, int n, int hf, int k0) {
  v16b f;
#pragma unroll
  for (int v = 0; v < 8; ++v) {
    int k = k0 + 16 * hf + 2 * v;
    f[2 * v]     = Src[(size_t)k * ld + n];
    f[2 * v + 1] = Src[(size_t)(k + 1) * ld + n];
  }
  return f;
}

__device__ __forceinline__ float gelu_tanh(float x) {
  float x3 = x * x * x;
  float t = 0.7978845608028654f * (x + 0.044715f * x3);
  return 0.5f * x * (1.0f + tanhf(t));
}

// ---------------- small elementwise kernels ----------------
__global__ __launch_bounds__(256) void silu_kernel(const float* __restrict__ v,
                                                   float* __restrict__ o) {
  int i = blockIdx.x * 256 + threadIdx.x;
  if (i < HID) { float x = v[i]; o[i] = x / (1.0f + __expf(-x)); }
}

// mod = W(12288x2048) @ silu(vec) + b   (one output per wave)
__global__ __launch_bounds__(256) void mod_gemv_kernel(const float* __restrict__ sv,
                                                       const float* __restrict__ w,
                                                       const float* __restrict__ b,
                                                       float* __restrict__ out) {
  int row = blockIdx.x * 8 + (threadIdx.x >> 5);
  int lane = threadIdx.x & 31;
  const float* wr = w + (size_t)row * HID;
  float s = 0.f;
  for (int j = lane; j < HID; j += 32) s += sv[j] * wr[j];
  s = wave_sum32(s);
  if (lane == 0) out[row] = s + b[row];
}

// layernorm over HID then (1+sc)*xn + sh, emit bf16 A-matrix rows
__global__ __launch_bounds__(256) void ln_mod_kernel(const float* __restrict__ x,
                                                     const float* __restrict__ sh,
                                                     const float* __restrict__ sc,
                                                     __bf16* __restrict__ out) {
  __shared__ float red[16];
  int row = blockIdx.x;
  int lane = threadIdx.x & 31, wv = threadIdx.x >> 5;
  const float* xr = x + (size_t)row * HID;
  float s = 0.f, ss = 0.f;
  for (int j = threadIdx.x; j < HID; j += 256) { float v = xr[j]; s += v; ss += v * v; }
  s = wave_sum32(s);
  ss = wave_sum32(ss);
  if (lane == 0) { red[wv] = s; red[8 + wv] = ss; }
  __syncthreads();
  float ts = 0.f, tss = 0.f;
#pragma unroll
  for (int i = 0; i < 8; ++i) { ts += red[i]; tss += red[8 + i]; }
  float mu = ts * (1.0f / HID);
  float var = tss * (1.0f / HID) - mu * mu;
  float rstd = rsqrtf(var + 1e-6f);
  for (int j = threadIdx.x; j < HID; j += 256) {
    float xn = (xr[j] - mu) * rstd;
    out[(size_t)row * HID + j] = f2b((1.0f + sc[j]) * xn + sh[j]);
  }
}

// cast kv caches to bf16 working copies
__global__ __launch_bounds__(256) void cache_cast_kernel(const float* __restrict__ kc,
                                                         const float* __restrict__ vc,
                                                         __bf16* __restrict__ kb,
                                                         __bf16* __restrict__ vb, int n) {
  for (int i = blockIdx.x * 256 + threadIdx.x; i < n; i += gridDim.x * 256) {
    kb[i] = f2b(kc[i]);
    vb[i] = f2b(vc[i]);
  }
}

// rmsnorm(q,k) + RoPE + scatter into bf16 caches; one (pos,head) per wave
__global__ __launch_bounds__(256) void qk_prep_kernel(
    const float* __restrict__ qkv, const float* __restrict__ pe,
    const float* __restrict__ iqn, const float* __restrict__ ikn,
    const float* __restrict__ tqn, const float* __restrict__ tkn,
    const long long* __restrict__ live,
    __bf16* __restrict__ qb, __bf16* __restrict__ kcb, __bf16* __restrict__ vcb) {
  int w = blockIdx.x * 8 + (threadIdx.x >> 5);
  if (w >= L_LIVE * NH) return;
  int lane = threadIdx.x & 31;
  int pos = w >> 4, hh = w & 15;
  const float* qn = (pos < L_TXT) ? tqn : iqn;
  const float* kn = (pos < L_TXT) ? tkn : ikn;
  const float* base = qkv + (size_t)pos * (3 * HID) + hh * HD;
  int d0 = lane * 4;
  float xq[4], xk[4], xv[4];
#pragma unroll
  for (int i = 0; i < 4; ++i) {
    xq[i] = base[d0 + i];
    xk[i] = base[HID + d0 + i];
    xv[i] = base[2 * HID + d0 + i];
  }
  float sq = 0.f, sk = 0.f;
#pragma unroll
  for (int i = 0; i < 4; ++i) { sq += xq[i] * xq[i]; sk += xk[i] * xk[i]; }
  sq = wave_sum32(sq);
  sk = wave_sum32(sk);
  float rq = rsqrtf(sq * (1.0f / HD) + 1e-6f);
  float rk = rsqrtf(sk * (1.0f / HD) + 1e-6f);
#pragma unroll
  for (int i = 0; i < 4; ++i) { xq[i] *= rq * qn[d0 + i]; xk[i] *= rk * kn[d0 + i]; }
  // RoPE: pe[pos][t][i][j]; out[t][i] = pe[t][i][0]*x[2t] + pe[t][i][1]*x[2t+1]
  const float* per = pe + (size_t)pos * (HD / 2) * 4;
  float oq[4], ok[4];
#pragma unroll
  for (int p = 0; p < 2; ++p) {
    int t = lane * 2 + p;
    const float* pt = per + t * 4;
    float a = xq[2 * p], b = xq[2 * p + 1];
    oq[2 * p]     = pt[0] * a + pt[1] * b;
    oq[2 * p + 1] = pt[2] * a + pt[3] * b;
    a = xk[2 * p]; b = xk[2 * p + 1];
    ok[2 * p]     = pt[0] * a + pt[1] * b;
    ok[2 * p + 1] = pt[2] * a + pt[3] * b;
  }
  long long r = live[pos];
#pragma unroll
  for (int i = 0; i < 4; ++i) {
    qb[(size_t)hh * L_LIVE * HD + (size_t)pos * HD + d0 + i] = f2b(oq[i]);
    kcb[(size_t)hh * L_FULL * HD + (size_t)r * HD + d0 + i]  = f2b(ok[i]);
    vcb[(size_t)hh * L_FULL * HD + (size_t)r * HD + d0 + i]  = f2b(xv[i]);
  }
}

// ---------------- WMMA GEMM: C = A(bf16, MxK) @ W(f32, NxK)^T + bias ----------------
// EPI 0: f32 out.  EPI 1: out = res + gate[col]*(acc+bias).  EPI 2: gelu -> bf16 out.
template <int EPI>
__global__ __launch_bounds__(256) void gemm_kernel(
    const __bf16* __restrict__ A, const float* __restrict__ W,
    const float* __restrict__ bias, float* __restrict__ outf,
    __bf16* __restrict__ outh, const float* __restrict__ res,
    const float* __restrict__ gate, int M, int N, int K) {
  const int wv = threadIdx.x >> 5, lane = threadIdx.x & 31;
  const int m = lane & 15, hf = lane >> 4;
  const int m0 = blockIdx.y * 64 + (wv >> 2) * 32;
  const int n0 = blockIdx.x * 64 + (wv & 3) * 16;
  const int n = n0 + m;
  v8f c0, c1;
#pragma unroll
  for (int i = 0; i < 8; ++i) { c0[i] = 0.f; c1[i] = 0.f; }
  for (int k0 = 0; k0 < K; k0 += 32) {
    // prefetch next k-step of the streamed fp32 weights (global_prefetch_b8)
    if (k0 + 32 < K) __builtin_prefetch(W + (size_t)n * K + k0 + 32 + 16 * hf, 0, 1);
    v16b b  = frag_b_nk_f32(W, K, n, hf, k0);
    v16b a0 = frag_a_bf16(A + (size_t)m0 * K, K, m, hf, k0);
    v16b a1 = frag_a_bf16(A + (size_t)(m0 + 16) * K, K, m, hf, k0);
    c0 = WMMA_BF16(a0, b, c0);
    c1 = WMMA_BF16(a1, b, c1);
  }
  float bc = bias[n];
#pragma unroll
  for (int r = 0; r < 8; ++r) {
    int row0 = m0 + r + 8 * hf;
    int row1 = row0 + 16;
    float v0 = c0[r] + bc, v1 = c1[r] + bc;
    if (EPI == 0) {
      outf[(size_t)row0 * N + n] = v0;
      outf[(size_t)row1 * N + n] = v1;
    } else if (EPI == 1) {
      float g = gate[n];
      outf[(size_t)row0 * N + n] = res[(size_t)row0 * N + n] + g * v0;
      outf[(size_t)row1 * N + n] = res[(size_t)row1 * N + n] + g * v1;
    } else {
      outh[(size_t)row0 * N + n] = f2b(gelu_tanh(v0));
      outh[(size_t)row1 * N + n] = f2b(gelu_tanh(v1));
    }
  }
}

// ---------------- flash attention (WMMA, online softmax) ----------------
// grid (L_LIVE/128, NH), block 256.  Each wave owns 16 query rows.
__global__ __launch_bounds__(256) void attn_kernel(
    const __bf16* __restrict__ q,   // [NH][L_LIVE][HD]
    const __bf16* __restrict__ kc,  // [NH][L_FULL][HD]
    const __bf16* __restrict__ vc,  // [NH][L_FULL][HD]
    __bf16* __restrict__ out) {     // [L_LIVE][HID]
  __shared__ __bf16 Pl[8][16][32];
  const int hh = blockIdx.y;
  const int wv = threadIdx.x >> 5, lane = threadIdx.x & 31;
  const int m = lane & 15, hf = lane >> 4;
  const int q0 = blockIdx.x * 128 + wv * 16;
  const __bf16* qh = q + (size_t)hh * L_LIVE * HD;
  const __bf16* kh = kc + (size_t)hh * L_FULL * HD;
  const __bf16* vh = vc + (size_t)hh * L_FULL * HD;
  const float scale = 0.08838834764831845f;  // 1/sqrt(128)

  v16b qf[4];
#pragma unroll
  for (int kk = 0; kk < 4; ++kk)
    qf[kk] = frag_a_bf16(qh + (size_t)q0 * HD, HD, m, hf, kk * 32);

  v8f O[8];
  float mrun[8], lrun[8];
#pragma unroll
  for (int r = 0; r < 8; ++r) { mrun[r] = -1e30f; lrun[r] = 0.f; }
#pragma unroll
  for (int dt = 0; dt < 8; ++dt)
#pragma unroll
    for (int r = 0; r < 8; ++r) O[dt][r] = 0.f;

  for (int it = 0; it < L_FULL / 32; ++it) {
    int kb = it * 32;
    v8f S0, S1;
#pragma unroll
    for (int i = 0; i < 8; ++i) { S0[i] = 0.f; S1[i] = 0.f; }
#pragma unroll
    for (int kk = 0; kk < 4; ++kk) {
      v16b b0 = frag_b_nk_bf16(kh, HD, kb + m, hf, kk * 32);
      S0 = WMMA_BF16(qf[kk], b0, S0);
      v16b b1 = frag_b_nk_bf16(kh, HD, kb + 16 + m, hf, kk * 32);
      S1 = WMMA_BF16(qf[kk], b1, S1);
    }
    // online softmax over the 32 new columns — reductions stay on the VALU
    // (DPP16 butterfly), co-executing with the matrix pipe; no DS round-trips.
#pragma unroll
    for (int r = 0; r < 8; ++r) {
      float s0 = S0[r] * scale, s1 = S1[r] * scale;
      float rm = row_max16(fmaxf(s0, s1));
      float mn = fmaxf(mrun[r], rm);
      float al = __expf(mrun[r] - mn);
      float p0 = __expf(s0 - mn), p1 = __expf(s1 - mn);
      float rs = row_sum16(p0 + p1);
      lrun[r] = lrun[r] * al + rs;
      mrun[r] = mn;
#pragma unroll
      for (int dt = 0; dt < 8; ++dt) O[dt][r] *= al;
      int row = r + 8 * hf;
      Pl[wv][row][m]      = f2b(p0);
      Pl[wv][row][16 + m] = f2b(p1);
    }
    // same-wave LDS write->read: DS ops are in order, just drain the counter
    asm volatile("s_wait_dscnt 0" ::: "memory");
    v16b pf = frag_a_bf16(&Pl[wv][0][0], 32, m, hf, 0);
#pragma unroll
    for (int dt = 0; dt < 8; ++dt) {
      v16b bv = frag_b_kn_bf16(vh + (size_t)kb * HD, HD, dt * 16 + m, hf, 0);
      O[dt] = WMMA_BF16(pf, bv, O[dt]);
    }
  }
#pragma unroll
  for (int r = 0; r < 8; ++r) {
    float inv = 1.0f / lrun[r];
    int row = q0 + r + 8 * hf;
#pragma unroll
    for (int dt = 0; dt < 8; ++dt)
      out[(size_t)row * HID + hh * HD + dt * 16 + m] = f2b(O[dt][r] * inv);
  }
}

// ---------------- host launch ----------------
extern "C" void kernel_launch(void* const* d_in, const int* in_sizes, int n_in,
                              void* d_out, int out_size, void* d_ws, size_t ws_size,
                              hipStream_t stream) {
  const float* img = (const float*)d_in[0];
  const float* txt = (const float*)d_in[1];
  const float* vec = (const float*)d_in[2];
  const float* pe  = (const float*)d_in[3];
  const float* k_cache = (const float*)d_in[4];
  const float* v_cache = (const float*)d_in[5];
  const long long* live = (const long long*)d_in[6];

  const float* p_mod_w[2]  = {(const float*)d_in[7],  (const float*)d_in[19]};
  const float* p_mod_b[2]  = {(const float*)d_in[8],  (const float*)d_in[20]};
  const float* p_qkv_w[2]  = {(const float*)d_in[9],  (const float*)d_in[21]};
  const float* p_qkv_b[2]  = {(const float*)d_in[10], (const float*)d_in[22]};
  const float* p_qnorm[2]  = {(const float*)d_in[11], (const float*)d_in[23]};
  const float* p_knorm[2]  = {(const float*)d_in[12], (const float*)d_in[24]};
  const float* p_proj_w[2] = {(const float*)d_in[13], (const float*)d_in[25]};
  const float* p_proj_b[2] = {(const float*)d_in[14], (const float*)d_in[26]};
  const float* p_mlp0_w[2] = {(const float*)d_in[15], (const float*)d_in[27]};
  const float* p_mlp0_b[2] = {(const float*)d_in[16], (const float*)d_in[28]};
  const float* p_mlp2_w[2] = {(const float*)d_in[17], (const float*)d_in[29]};
  const float* p_mlp2_b[2] = {(const float*)d_in[18], (const float*)d_in[30]};
  // index 0 = img, 1 = txt

  // ---- workspace carve-out ----
  char* w = (char*)d_ws;
  auto take = [&](size_t bytes) { char* p = w; w += (bytes + 255) & ~(size_t)255; return p; };
  float* silu = (float*)take(HID * 4);
  float* modi = (float*)take(6 * HID * 4);
  float* modt = (float*)take(6 * HID * 4);
  __bf16* xln = (__bf16*)take((size_t)L_LIVE * HID * 2);
  float* qkv  = (float*)take((size_t)L_LIVE * 3 * HID * 4);  // reused as gelu-hidden (bf16)
  __bf16* qb  = (__bf16*)take((size_t)NH * L_LIVE * HD * 2);
  __bf16* kcb = (__bf16*)take((size_t)NH * L_FULL * HD * 2);
  __bf16* vcb = (__bf16*)take((size_t)NH * L_FULL * HD * 2);
  __bf16* attnb = (__bf16*)take((size_t)L_LIVE * HID * 2);
  float* x1   = (float*)take((size_t)L_LIVE * HID * 4);
  __bf16* hg  = (__bf16*)qkv;  // [L_LIVE][MLPD] bf16, fits inside dead qkv region

  float* out_img = (float*)d_out;                         // [L_IMG][HID]
  float* out_txt = (float*)d_out + (size_t)L_IMG * HID;   // [L_TXT][HID]

  // 1) modulation vectors
  silu_kernel<<<(HID + 255) / 256, 256, 0, stream>>>(vec, silu);
  mod_gemv_kernel<<<6 * HID / 8, 256, 0, stream>>>(silu, p_mod_w[0], p_mod_b[0], modi);
  mod_gemv_kernel<<<6 * HID / 8, 256, 0, stream>>>(silu, p_mod_w[1], p_mod_b[1], modt);

  // 2) layernorm + (1+sc1)*x + sh1 -> bf16 A-matrix (txt rows 0..255, img rows 256..)
  ln_mod_kernel<<<L_TXT, 256, 0, stream>>>(txt, modt + 0, modt + HID, xln);
  ln_mod_kernel<<<L_IMG, 256, 0, stream>>>(img, modi + 0, modi + HID, xln + (size_t)L_TXT * HID);

  // 3) QKV projections (WMMA)
  gemm_kernel<0><<<dim3(3 * HID / 64, L_TXT / 64), 256, 0, stream>>>(
      xln, p_qkv_w[1], p_qkv_b[1], qkv, nullptr, nullptr, nullptr, L_TXT, 3 * HID, HID);
  gemm_kernel<0><<<dim3(3 * HID / 64, L_IMG / 64), 256, 0, stream>>>(
      xln + (size_t)L_TXT * HID, p_qkv_w[0], p_qkv_b[0], qkv + (size_t)L_TXT * 3 * HID,
      nullptr, nullptr, nullptr, L_IMG, 3 * HID, HID);

  // 4) bf16 KV-cache working copies, then rmsnorm+RoPE+scatter
  cache_cast_kernel<<<2048, 256, 0, stream>>>(k_cache, v_cache, kcb, vcb, NH * L_FULL * HD);
  qk_prep_kernel<<<(L_LIVE * NH) / 8, 256, 0, stream>>>(
      qkv, pe, p_qnorm[0], p_knorm[0], p_qnorm[1], p_knorm[1], live, qb, kcb, vcb);

  // 5) flash attention (WMMA), writes bf16 attn [L_LIVE][HID]
  attn_kernel<<<dim3(L_LIVE / 128, NH), 256, 0, stream>>>(qb, kcb, vcb, attnb);

  // 6) out-proj with gated residual -> x1 (f32)
  gemm_kernel<1><<<dim3(HID / 64, L_TXT / 64), 256, 0, stream>>>(
      attnb, p_proj_w[1], p_proj_b[1], x1, nullptr, txt, modt + 2 * HID, L_TXT, HID, HID);
  gemm_kernel<1><<<dim3(HID / 64, L_IMG / 64), 256, 0, stream>>>(
      attnb + (size_t)L_TXT * HID, p_proj_w[0], p_proj_b[0], x1 + (size_t)L_TXT * HID,
      nullptr, img, modi + 2 * HID, L_IMG, HID, HID);

  // 7) second layernorm + mod2 -> bf16 A-matrix (reuse xln)
  ln_mod_kernel<<<L_TXT, 256, 0, stream>>>(x1, modt + 3 * HID, modt + 4 * HID, xln);
  ln_mod_kernel<<<L_IMG, 256, 0, stream>>>(x1 + (size_t)L_TXT * HID, modi + 3 * HID,
                                           modi + 4 * HID, xln + (size_t)L_TXT * HID);

  // 8) MLP up-proj with fused gelu -> bf16 hidden
  gemm_kernel<2><<<dim3(MLPD / 64, L_TXT / 64), 256, 0, stream>>>(
      xln, p_mlp0_w[1], p_mlp0_b[1], nullptr, hg, nullptr, nullptr, L_TXT, MLPD, HID);
  gemm_kernel<2><<<dim3(MLPD / 64, L_IMG / 64), 256, 0, stream>>>(
      xln + (size_t)L_TXT * HID, p_mlp0_w[0], p_mlp0_b[0], nullptr,
      hg + (size_t)L_TXT * MLPD, nullptr, nullptr, L_IMG, MLPD, HID);

  // 9) MLP down-proj with gated residual, straight into d_out
  gemm_kernel<1><<<dim3(HID / 64, L_TXT / 64), 256, 0, stream>>>(
      hg, p_mlp2_w[1], p_mlp2_b[1], out_txt, nullptr, x1, modt + 5 * HID, L_TXT, HID, MLPD);
  gemm_kernel<1><<<dim3(HID / 64, L_IMG / 64), 256, 0, stream>>>(
      hg + (size_t)L_TXT * MLPD, p_mlp2_w[0], p_mlp2_b[0], out_img, nullptr,
      x1 + (size_t)L_TXT * HID, modi + 5 * HID, L_IMG, HID, MLPD);
}
